// CIN_15968688406979
// MI455X (gfx1250) — matrix operand order repaired and checked by reading, not compiled, not run on hardware
//
#include <hip/hip_runtime.h>

// ---------------- types ----------------
typedef __bf16 bf16_t;
typedef bf16_t v16bf __attribute__((ext_vector_type(16)));
typedef bf16_t v2bf  __attribute__((ext_vector_type(2)));
typedef float  v8f   __attribute__((ext_vector_type(8)));
typedef float  v2f   __attribute__((ext_vector_type(2)));
typedef unsigned int v4u __attribute__((ext_vector_type(4)));

// ---------------- problem constants ----------------
#define NUM_FIELD 39
#define FP        40          // padded field count
#define EMBED     16
#define H0        128         // CIN layer sizes
#define H1        128
#define BATCH     2048
#define KP1       (FP*FP)     // 1600 padded K for layer 1
#define KP2       (H0*FP)     // 5120 padded K for layer 2
#define ISTRIP    8           // i-rows per LDS strip
#define KSTRIP    (ISTRIP*FP) // 320 K-values per strip
#define CHUNKS_PER_STRIP (KSTRIP/32)   // 10
#define SSTRIDE   328         // ushort stride per d-row in strip (bank spread)
#define BUFSZ     (EMBED*SSTRIDE)      // 5248 ushorts per strip buffer
#define ETS       42          // Et row stride (floats)
#define XTS       129         // X1t row stride (floats)

// fp32 -> bf16 (round to nearest even), scalar path for the prep kernel
static __device__ inline unsigned short f2bf(float f) {
    unsigned u = __float_as_uint(f);
    unsigned r = u + 0x7fffu + ((u >> 16) & 1u);
    return (unsigned short)(r >> 16);
}

// pack two fp32 into one dword of bf16x2.
// Vector fptrunc: lowers to native v_cvt_pk_bf16_f32 when the backend supports
// BF16 conversion instructions (gfx950+/gfx1250), else expands to RNE bit math.
static __device__ inline unsigned pack_bf16(float a, float b) {
#if __has_builtin(__builtin_amdgcn_cvt_pk_bf16_f32)
    union { v2bf v; unsigned u; } c;
    c.v = __builtin_amdgcn_cvt_pk_bf16_f32(a, b);
    return c.u;
#else
    v2f f;
    f.x = a; f.y = b;
    union { v2bf v; unsigned u; } c;
    c.v = __builtin_convertvector(f, v2bf);
    return c.u;
#endif
}

static __device__ inline v16bf frag_cat(v4u a, v4u b) {
    union { v4u q[2]; v16bf v; } u;
    u.q[0] = a; u.q[1] = b;
    return u.v;
}

// ---------------- prep: repack W0/W1 fp32 -> padded bf16 ----------------
__global__ __launch_bounds__(256) void cin_repack_w(
        const float* __restrict__ W0, const float* __restrict__ W1,
        unsigned short* __restrict__ Wp0, unsigned short* __restrict__ Wp1) {
    int idx = blockIdx.x * 256 + threadIdx.x;
    const int n0 = H0 * KP1;                 // 204800
    const int n1 = H1 * KP2;                 // 655360
    if (idx < n0) {
        int h = idx / KP1, k = idx % KP1;
        int i = k / FP, j = k % FP;
        float v = (i < NUM_FIELD && j < NUM_FIELD)
                    ? W0[h * (NUM_FIELD * NUM_FIELD) + i * NUM_FIELD + j] : 0.f;
        Wp0[idx] = f2bf(v);
    } else if (idx < n0 + n1) {
        int e = idx - n0;
        int h = e / KP2, k = e % KP2;
        int i = k / FP, j = k % FP;
        float v = (j < NUM_FIELD)
                    ? W1[h * (H0 * NUM_FIELD) + i * NUM_FIELD + j] : 0.f;
        Wp1[e] = f2bf(v);
    }
}

// ---------------- main fused CIN kernel: one block per batch ----------------
__global__ __launch_bounds__(256) void cin_main(
        const float* __restrict__ x_emb,
        const unsigned short* __restrict__ Wp0,
        const unsigned short* __restrict__ Wp1,
        float* __restrict__ out) {

    __shared__ __align__(16) float Et [EMBED * ETS];       // e[j,d] transposed -> [d][j]
    __shared__ __align__(16) float X1t[EMBED * XTS];       // layer-1 out       -> [d][h]
    __shared__ __align__(16) unsigned short Pb[2 * BUFSZ]; // double-buffered strips [d][k]

    const int b    = blockIdx.x;
    const int t    = threadIdx.x;
    const int lane = t & 31;
    const int wave = t >> 5;

    // -------- load E transposed into LDS, zero the j=39 pad column --------
    const float* E = x_emb + (long)b * NUM_FIELD * EMBED;
    for (int e = t; e < NUM_FIELD * EMBED; e += 256) {
        int j = e >> 4, d = e & 15;
        Et[d * ETS + j] = E[e];
    }
    if (t < EMBED) Et[t * ETS + (FP - 1)] = 0.f;
    __syncthreads();

    // -------- strip builder: all 256 threads; 16 threads per d-row, 10 k-pairs each
    //          P[d][kl] = Xt[d][i0 + kl/40] * Et[d][kl%40]  (bf16 pairs)
    const int bld_d = t >> 4;       // 0..15 : embedding channel
    const int bld_p = t & 15;       // pair slot within the d-row
    auto build = [&](const float* __restrict__ Xt, int xts, int i0,
                     unsigned short* __restrict__ dst) {
        const float* xrow = Xt + bld_d * xts + i0;
        const float* erow = Et + bld_d * ETS;
        unsigned short* drow = dst + bld_d * SSTRIDE;
#pragma unroll
        for (int m = 0; m < KSTRIP / 32; ++m) {
            int kl = 2 * (bld_p + 16 * m);   // even local k in [0, 320)
            int i  = kl / FP;                // 0..7  (same i for kl, kl+1)
            int j  = kl % FP;
            float xv = xrow[i];
            unsigned pk = pack_bf16(xv * erow[j], xv * erow[j + 1]);
            *(unsigned*)&drow[kl] = pk;
        }
    };

    // -------- per-lane WMMA fragment geometry (wave32 layouts) --------
    const int mrow   = lane & 15;                 // A row index within h-tile
    const int kbaseA = (lane < 16) ? 0 : 8;       // A: halves at +0 and +16
    const int kbaseB = (lane < 16) ? 0 : 8;       // B: halves at +0 and +16
    const int dcol   = lane & 15;                 // N = embedding channel

    // ================= layer 1: X1 = W0 (x) [e ⊗ e] =================
    v8f acc = {0.f, 0.f, 0.f, 0.f, 0.f, 0.f, 0.f, 0.f};
    {
        const unsigned short* Wrow = Wp0 + (wave * 16 + mrow) * KP1;
        const int NSTRIP = FP / ISTRIP;           // 5
        build(Et, ETS, 0, Pb);
        __syncthreads();
        for (int s = 0; s < NSTRIP; ++s) {
            const unsigned short* cur = Pb + (s & 1) * BUFSZ;
            unsigned short*       nxt = Pb + ((s + 1) & 1) * BUFSZ;
            if (s + 1 < NSTRIP) build(Et, ETS, (s + 1) * ISTRIP, nxt);
            const unsigned short* wk = Wrow + s * KSTRIP;
#pragma unroll
            for (int c = 0; c < CHUNKS_PER_STRIP; ++c) {
                int ka = c * 32 + kbaseA;
                v4u a0 = *(const v4u*)(wk + ka);
                v4u a1 = *(const v4u*)(wk + ka + 16);
                const unsigned short* bp = cur + dcol * SSTRIDE + c * 32 + kbaseB;
                v4u b0 = *(const v4u*)(bp);
                v4u b1 = *(const v4u*)(bp + 16);
                acc = __builtin_amdgcn_wmma_f32_16x16x32_bf16(
                        false, frag_cat(a0, a1),
                        false, frag_cat(b0, b1),
                        (short)0, acc, false, false);
            }
            __syncthreads();
        }
    }

    // scatter D-frag -> X1t[d][h]  (D layout: VGPR r -> M = r (+8 for hi lanes))
#pragma unroll
    for (int r = 0; r < 8; ++r) {
        int h = wave * 16 + r + ((lane < 16) ? 0 : 8);
        X1t[dcol * XTS + h] = acc[r];
    }
    __syncthreads();

    // out0[b,h] = sum_d X1[h,d]
    if (t < H0) {
        float s = 0.f;
#pragma unroll
        for (int d = 0; d < EMBED; ++d) s += X1t[d * XTS + t];
        out[(long)b * (H0 + H1) + t] = s;
    }

    // ================= layer 2: X2 = W1 (x) [x1 ⊗ e] =================
    v8f acc2 = {0.f, 0.f, 0.f, 0.f, 0.f, 0.f, 0.f, 0.f};
    {
        const unsigned short* Wrow = Wp1 + (wave * 16 + mrow) * KP2;
        const int NSTRIP = H0 / ISTRIP;           // 16
        build(X1t, XTS, 0, Pb);
        __syncthreads();
        for (int s = 0; s < NSTRIP; ++s) {
            const unsigned short* cur = Pb + (s & 1) * BUFSZ;
            unsigned short*       nxt = Pb + ((s + 1) & 1) * BUFSZ;
            if (s + 1 < NSTRIP) build(X1t, XTS, (s + 1) * ISTRIP, nxt);
            const unsigned short* wk = Wrow + s * KSTRIP;
#pragma unroll
            for (int c = 0; c < CHUNKS_PER_STRIP; ++c) {
                int ka = c * 32 + kbaseA;
                v4u a0 = *(const v4u*)(wk + ka);
                v4u a1 = *(const v4u*)(wk + ka + 16);
                const unsigned short* bp = cur + dcol * SSTRIDE + c * 32 + kbaseB;
                v4u b0 = *(const v4u*)(bp);
                v4u b1 = *(const v4u*)(bp + 16);
                acc2 = __builtin_amdgcn_wmma_f32_16x16x32_bf16(
                        false, frag_cat(a0, a1),
                        false, frag_cat(b0, b1),
                        (short)0, acc2, false, false);
            }
            __syncthreads();
        }
    }

    // reuse X1t for the final channel-sum (all strip reads are done)
#pragma unroll
    for (int r = 0; r < 8; ++r) {
        int h = wave * 16 + r + ((lane < 16) ? 0 : 8);
        X1t[dcol * XTS + h] = acc2[r];
    }
    __syncthreads();

    if (t < H1) {
        float s = 0.f;
#pragma unroll
        for (int d = 0; d < EMBED; ++d) s += X1t[d * XTS + t];
        out[(long)b * (H0 + H1) + H0 + t] = s;
    }
}

// ---------------- host launcher ----------------
extern "C" void kernel_launch(void* const* d_in, const int* in_sizes, int n_in,
                              void* d_out, int out_size, void* d_ws, size_t ws_size,
                              hipStream_t stream) {
    const float* x_emb = (const float*)d_in[0];
    const float* W0    = (const float*)d_in[1];
    const float* W1    = (const float*)d_in[2];
    float*       out   = (float*)d_out;

    unsigned short* Wp0 = (unsigned short*)d_ws;            // 128*1600 bf16
    unsigned short* Wp1 = Wp0 + (size_t)H0 * KP1;           // 128*5120 bf16
    // total ws use: (204800 + 655360) * 2 = 1,720,320 bytes

    const int prep_elems = H0 * KP1 + H1 * KP2;             // 860160
    cin_repack_w<<<(prep_elems + 255) / 256, 256, 0, stream>>>(W0, W1, Wp0, Wp1);
    cin_main<<<BATCH, 256, 0, stream>>>(x_emb, Wp0, Wp1, out);
}
